// AqtDotGeneral_36344013259233
// MI455X (gfx1250) — 8-bit matrix operand order repaired and checked by reading, compiled, not run on hardware
//
#include <hip/hip_runtime.h>
#include <stdint.h>

typedef __attribute__((ext_vector_type(8))) int v8i;

#define QMAX_F 127.0f

__global__ void rhs_quant_kernel(const float* __restrict__ rhs,
                                 int8_t* __restrict__ qfrag,
                                 float* __restrict__ scales)
{
    const int f = blockIdx.x * blockDim.x + threadIdx.x;
    if (f >= 512) return;

    float amax = 0.0f;
    for (int d = 0; d < 512; ++d)
        amax = fmaxf(amax, fabsf(rhs[d * 512 + f]));

    const float scale = (amax > 0.0f) ? (amax / QMAX_F) : 1.0f;
    const float inv   = (amax > 0.0f) ? (QMAX_F / amax) : 0.0f;
    scales[f] = scale;

    const int ntile   = f >> 4;
    const int lane_lo = f & 15;
    for (int d = 0; d < 512; ++d) {
        const float x = rhs[d * 512 + f];
        int q = (int)rintf(x * inv);
        q = q < -127 ? -127 : (q > 127 ? 127 : q);

        const int kc   = d >> 6;
        const int kr   = d & 63;
        const int gi   = kr >> 4;
        const int vg   = ((gi & 2) << 1) + ((kr & 15) >> 2);
        const int lane = lane_lo | ((gi & 1) << 4);
        const int byt  = kr & 3;
        qfrag[(((ntile << 3) + kc) << 10) + lane * 32 + vg * 4 + byt] = (int8_t)q;
    }
}

__device__ __forceinline__ int quant_pack4(float4 x, float inv) {
    int q0 = (int)rintf(x.x * inv);
    int q1 = (int)rintf(x.y * inv);
    int q2 = (int)rintf(x.z * inv);
    int q3 = (int)rintf(x.w * inv);
    q0 = q0 < -127 ? -127 : (q0 > 127 ? 127 : q0);
    q1 = q1 < -127 ? -127 : (q1 > 127 ? 127 : q1);
    q2 = q2 < -127 ? -127 : (q2 > 127 ? 127 : q2);
    q3 = q3 < -127 ? -127 : (q3 > 127 ? 127 : q3);
    return (q0 & 255) | ((q1 & 255) << 8) | ((q2 & 255) << 16) | ((q3 & 255) << 24);
}

__device__ __forceinline__ v8i load_bfrag(const int8_t* p) {
    const int4 b0 = *(const int4*)(p);
    const int4 b1 = *(const int4*)(p + 16);
    v8i b;
    b[0] = b0.x; b[1] = b0.y; b[2] = b0.z; b[3] = b0.w;
    b[4] = b1.x; b[5] = b1.y; b[6] = b1.z; b[7] = b1.w;
    return b;
}

__global__ __launch_bounds__(256) void aqt_gemm_kernel(
    const float*  __restrict__ lhs,
    const int8_t* __restrict__ qfrag,
    const float*  __restrict__ rscale,
    float*        __restrict__ out)
{
    __shared__ __attribute__((aligned(16))) int8_t lds_b[256 * 1024];

    const int lane  = threadIdx.x & 31;
    const int wave  = threadIdx.x >> 5;
    const int mtile = blockIdx.x * 8 + wave;

    {
        const unsigned tid   = threadIdx.x;
        const uint64_t gbase = (uint64_t)(uintptr_t)qfrag;
        const unsigned lds_base = (unsigned)(uintptr_t)
            (__attribute__((address_space(3))) int8_t*)lds_b;
#pragma unroll 8
        for (int i = 0; i < 64; ++i) {
            const unsigned off  = (i * 256u + tid) * 16u;
            const unsigned ldsa = lds_base + off;
            const uint64_t ga   = gbase + off;
            asm volatile("global_load_async_to_lds_b128 %0, %1, off"
                         :: "v"(ldsa), "v"(ga) : "memory");
        }
        asm volatile("s_wait_asynccnt 0x0" ::: "memory");
        __syncthreads();
    }

    const int m = lane & 15;
    const int h = lane >> 4;

    const float* rp = lhs + (size_t)(mtile * 16 + m) * 512;

    float amax = 0.0f;
#pragma unroll
    for (int g = 0; g < 32; ++g) {
        const int k = g * 16 + h * 8;
        const float4 x0 = *(const float4*)(rp + k);
        const float4 x1 = *(const float4*)(rp + k + 4);
        amax = fmaxf(amax, fmaxf(fmaxf(fabsf(x0.x), fabsf(x0.y)),
                                 fmaxf(fabsf(x0.z), fabsf(x0.w))));
        amax = fmaxf(amax, fmaxf(fmaxf(fabsf(x1.x), fabsf(x1.y)),
                                 fmaxf(fabsf(x1.z), fabsf(x1.w))));
    }
    amax = fmaxf(amax, __shfl_xor(amax, 16, 32));
    const float lscale = (amax > 0.0f) ? (amax / QMAX_F) : 1.0f;
    const float inv    = (amax > 0.0f) ? (QMAX_F / amax) : 0.0f;

    v8i a[8];
#pragma unroll
    for (int kc = 0; kc < 8; ++kc) {
#pragma unroll
        for (int j = 0; j < 4; ++j) {
            const int k = kc * 64 + (j >> 1) * 32 + h * 16 + (j & 1) * 8;
            const float4 x0 = *(const float4*)(rp + k);
            const float4 x1 = *(const float4*)(rp + k + 4);
            a[kc][2 * j]     = quant_pack4(x0, inv);
            a[kc][2 * j + 1] = quant_pack4(x1, inv);
        }
    }

    float lsr[8];
#pragma unroll
    for (int r = 0; r < 8; ++r)
        lsr[r] = __shfl(lscale, r + 8 * h, 32);

    for (int nt = 0; nt < 32; nt += 2) {
        const int8_t* bp0 = lds_b + ((size_t)nt << 13) + lane * 32;
        const int8_t* bp1 = bp0 + 8192;
        v8i acc0 = {};
        v8i acc1 = {};
#pragma unroll
        for (int kc = 0; kc < 8; ++kc) {
            const v8i b0 = load_bfrag(bp0 + (kc << 10));
            const v8i b1 = load_bfrag(bp1 + (kc << 10));
            acc0 = __builtin_amdgcn_wmma_i32_16x16x64_iu8(
                true, a[kc], true, b0, acc0, false, false);
            acc1 = __builtin_amdgcn_wmma_i32_16x16x64_iu8(
                true, a[kc], true, b1, acc1, false, false);
        }

        const float fs0 = rscale[nt * 16 + m];
        const float fs1 = rscale[nt * 16 + 16 + m];
        float* op0 = out + (size_t)(mtile * 16 + 8 * h) * 512 + nt * 16 + m;
#pragma unroll
        for (int r = 0; r < 8; ++r) {
            op0[(size_t)r * 512]      = (float)acc0[r] * lsr[r] * fs0;
            op0[(size_t)r * 512 + 16] = (float)acc1[r] * lsr[r] * fs1;
        }
    }
}

extern "C" void kernel_launch(void* const* d_in, const int* in_sizes, int n_in,
                              void* d_out, int out_size, void* d_ws, size_t ws_size,
                              hipStream_t stream) {
    const float* lhs = (const float*)d_in[0];
    const float* rhs = (const float*)d_in[1];
    float* out       = (float*)d_out;

    int8_t* qfrag  = (int8_t*)d_ws;
    float*  rscale = (float*)((char*)d_ws + 256 * 1024);

    rhs_quant_kernel<<<2, 256, 0, stream>>>(rhs, qfrag, rscale);

    aqt_gemm_kernel<<<512, 256, 0, stream>>>(lhs, qfrag, rscale, out);
}
